// PES_78950088835328
// MI455X (gfx1250) — compile-verified, matrix operand order checked by reading, not run on hardware
//
#include <hip/hip_runtime.h>

#define NWAVE 12
#define NL 13
#define NFEAT 36
#define PI_OVER_CUT 0.52359877559829887308f /* pi / 6.0 */

typedef __attribute__((ext_vector_type(16))) __bf16 v16bf;
typedef __attribute__((ext_vector_type(8)))  float  v8f;

// fast sigmoid: v_exp_f32 + v_add + v_rcp_f32 (avoids IEEE divide expansion)
__device__ __forceinline__ float sigf(float x) {
    return __builtin_amdgcn_rcpf(1.0f + __expf(-x));
}

// ---------------------------------------------------------------------------
// Templated WMMA GEMM: Y[R,O] (=|+=) act(X[R,K]) @ B + bias
//   TRANSB=false: B(k,n) = W[k*O + n]   (forward,   W is [K,O])
//   TRANSB=true : B(k,n) = W[n*K + k]   (backward,  W is [O,K] accessed transposed)
//   HASACT: A elements become aA[k]*silu(gA[k]*x) (fused consumer-side activation)
// One wave computes a 16 x (CT*16) strip: one A fragment feeds CT back-to-back
// v_wmma issues per K-step (A traffic / CT, wmma issue overlapped with B loads).
// A fragment lane(hi,lo): row=lo, element e -> k = k0 + hi*8 + e (e<8),
//                                            k = k0 + 16 + hi*8 + (e-8) (e>=8).
// ---------------------------------------------------------------------------
template<int K, int O, int CT, bool TRANSB, bool HASACT>
__global__ __launch_bounds__(32)
void eann_gemm_t(const float* __restrict__ X,
                 const float* __restrict__ aA, const float* __restrict__ gA,
                 const float* __restrict__ W,
                 const float* __restrict__ bias,
                 float* __restrict__ Y, int accumulate)
{
    constexpr int  KT = (K + 31) & ~31;
    constexpr bool KP = (K % 32) != 0;     // K needs zero padding
    constexpr bool OB = (O % 16) != 0;     // O needs column bounds
    const int lane = threadIdx.x & 31;
    const int lo = lane & 15;
    const int hi = lane >> 4;
    const int row0 = blockIdx.x << 4;
    const int colbase = blockIdx.y * (CT * 16);
    const float* xrow = X + (size_t)(row0 + lo) * K;

    v8f acc[CT];
#pragma unroll
    for (int c = 0; c < CT; ++c) acc[c] = (v8f){0.f, 0.f, 0.f, 0.f, 0.f, 0.f, 0.f, 0.f};

#pragma unroll
    for (int k0 = 0; k0 < KT; k0 += 32) {
        const int ka = k0 + hi * 8;
        float xv[16];

        // ---------------- A fragment (shared across CT column tiles) --------
        if constexpr (!KP) {
            const float4 x0 = *(const float4*)(xrow + ka);
            const float4 x1 = *(const float4*)(xrow + ka + 4);
            const float4 x2 = *(const float4*)(xrow + ka + 16);
            const float4 x3 = *(const float4*)(xrow + ka + 20);
            xv[0]=x0.x;  xv[1]=x0.y;  xv[2]=x0.z;  xv[3]=x0.w;
            xv[4]=x1.x;  xv[5]=x1.y;  xv[6]=x1.z;  xv[7]=x1.w;
            xv[8]=x2.x;  xv[9]=x2.y;  xv[10]=x2.z; xv[11]=x2.w;
            xv[12]=x3.x; xv[13]=x3.y; xv[14]=x3.z; xv[15]=x3.w;
            if constexpr (HASACT) {
                const float4 a0 = *(const float4*)(aA + ka);
                const float4 a1 = *(const float4*)(aA + ka + 4);
                const float4 a2 = *(const float4*)(aA + ka + 16);
                const float4 a3 = *(const float4*)(aA + ka + 20);
                const float4 g0 = *(const float4*)(gA + ka);
                const float4 g1 = *(const float4*)(gA + ka + 4);
                const float4 g2 = *(const float4*)(gA + ka + 16);
                const float4 g3 = *(const float4*)(gA + ka + 20);
                const float av[16] = {a0.x,a0.y,a0.z,a0.w, a1.x,a1.y,a1.z,a1.w,
                                      a2.x,a2.y,a2.z,a2.w, a3.x,a3.y,a3.z,a3.w};
                const float gv[16] = {g0.x,g0.y,g0.z,g0.w, g1.x,g1.y,g1.z,g1.w,
                                      g2.x,g2.y,g2.z,g2.w, g3.x,g3.y,g3.z,g3.w};
#pragma unroll
                for (int e = 0; e < 16; ++e) {
                    const float gz = gv[e] * xv[e];
                    xv[e] = av[e] * gz * sigf(gz);
                }
            }
        } else {
            // padded-K path (K=36 input layer): branchless clamped loads
#pragma unroll
            for (int e = 0; e < 16; ++e) {
                const int kk = k0 + ((e < 8) ? (hi * 8 + e) : (16 + hi * 8 + (e - 8)));
                const int kks = (kk < K) ? kk : 0;
                float a = xrow[kks];
                if constexpr (HASACT) { const float gz = gA[kks] * a; a = aA[kks] * gz * sigf(gz); }
                xv[e] = (kk < K) ? a : 0.f;
            }
        }
        v16bf fa;
#pragma unroll
        for (int e = 0; e < 16; ++e) fa[e] = (__bf16)xv[e];

        // ---------------- CT column tiles: B fragment + wmma ----------------
#pragma unroll
        for (int c = 0; c < CT; ++c) {
            const int col = colbase + c * 16 + lo;
            const int colc = OB ? (col < O ? col : 0) : col;
            float wv[16];
            if constexpr (!KP) {
                if constexpr (TRANSB) {
                    const float* wrow = W + (size_t)colc * K;
                    const float4 w0 = *(const float4*)(wrow + ka);
                    const float4 w1 = *(const float4*)(wrow + ka + 4);
                    const float4 w2 = *(const float4*)(wrow + ka + 16);
                    const float4 w3 = *(const float4*)(wrow + ka + 20);
                    wv[0]=w0.x;  wv[1]=w0.y;  wv[2]=w0.z;  wv[3]=w0.w;
                    wv[4]=w1.x;  wv[5]=w1.y;  wv[6]=w1.z;  wv[7]=w1.w;
                    wv[8]=w2.x;  wv[9]=w2.y;  wv[10]=w2.z; wv[11]=w2.w;
                    wv[12]=w3.x; wv[13]=w3.y; wv[14]=w3.z; wv[15]=w3.w;
                } else {
#pragma unroll
                    for (int e = 0; e < 16; ++e) {
                        const int kk = k0 + ((e < 8) ? (hi * 8 + e) : (16 + hi * 8 + (e - 8)));
                        wv[e] = W[(size_t)kk * O + colc];   // coalesced across lanes
                    }
                }
                if constexpr (OB) {
                    if (col >= O) {
#pragma unroll
                        for (int e = 0; e < 16; ++e) wv[e] = 0.f;
                    }
                }
            } else {
#pragma unroll
                for (int e = 0; e < 16; ++e) {
                    const int kk = k0 + ((e < 8) ? (hi * 8 + e) : (16 + hi * 8 + (e - 8)));
                    const int kks = (kk < K) ? kk : 0;
                    float b;
                    if constexpr (TRANSB) b = W[(size_t)colc * K + kks];
                    else                  b = W[(size_t)kks * O + colc];
                    const bool ok = (kk < K) && (!OB || col < O);
                    wv[e] = ok ? b : 0.f;
                }
            }
            v16bf fb;
#pragma unroll
            for (int e = 0; e < 16; ++e) fb[e] = (__bf16)wv[e];
            acc[c] = __builtin_amdgcn_wmma_f32_16x16x32_bf16(false, fa, false, fb,
                                                             (short)0, acc[c], false, false);
        }
    }

    // ---------------- epilogue ----------------
#pragma unroll
    for (int c = 0; c < CT; ++c) {
        const int col = colbase + c * 16 + lo;
        if (!OB || col < O) {
            const float bb = bias ? bias[col] : 0.f;
#pragma unroll
            for (int v = 0; v < 8; ++v) {
                const int row = row0 + v + 8 * hi;
                const float val = acc[c][v] + bb;
                if (accumulate) Y[(size_t)row * O + col] += val;
                else            Y[(size_t)row * O + col]  = val;
            }
        }
    }
}

// ---------------------------------------------------------------------------
// Pair forward: one block per center atom i; KN (=48) neighbor pairs reduced
// through LDS into B[i, 13, 12].  No global atomics, B fully overwritten.
// ---------------------------------------------------------------------------
__global__ void eann_pair_forward(const float* __restrict__ cart,
                                  const float* __restrict__ shifts,
                                  const float* __restrict__ rs,
                                  const float* __restrict__ inta,
                                  const float* __restrict__ cpar,
                                  const int* __restrict__ species,
                                  const int* __restrict__ neigh,
                                  int M, int KN,
                                  float* __restrict__ B)
{
    __shared__ float angS[48 * NL];
    __shared__ float gS[48 * NWAVE];
    const int i = blockIdx.x;
    const int p = threadIdx.x;

    if (p < KN) {
        const int m = i * KN + p;
        const int j = neigh[M + m];
        float dv[3], d2 = 0.f;
#pragma unroll
        for (int c = 0; c < 3; ++c) {
            dv[c] = cart[i * 3 + c] - cart[j * 3 + c] + shifts[m * 3 + c];
            d2 += dv[c] * dv[c];
        }
        const float dist = sqrtf(d2);
        const float cc = 0.5f * __cosf(dist * PI_OVER_CUT) + 0.5f;
        const float fc = cc * cc;
        float* ang = &angS[p * NL];
        ang[0] = fc;
#pragma unroll
        for (int c = 0; c < 3; ++c) ang[1 + c] = fc * dv[c];
#pragma unroll
        for (int a = 0; a < 3; ++a)
#pragma unroll
            for (int b = 0; b < 3; ++b) ang[4 + 3 * a + b] = fc * dv[a] * dv[b];
        const int sj = species[j];
#pragma unroll
        for (int w = 0; w < NWAVE; ++w) {
            const float dr = dist - rs[sj * NWAVE + w];
            gS[p * NWAVE + w] = __expf(inta[sj * NWAVE + w] * dr * dr) * cpar[sj * NWAVE + w];
        }
    }
    __syncthreads();
    if (p < NL * NWAVE) {
        const int l = p / NWAVE, w = p % NWAVE;
        float s = 0.f;
        for (int q = 0; q < KN; ++q) s += angS[q * NL + l] * gS[q * NWAVE + w];
        B[((size_t)i * NL + l) * NWAVE + w] = s;
    }
}

__global__ void eann_density(const float* __restrict__ B, float* __restrict__ dens, int N)
{
    const int idx = blockIdx.x * blockDim.x + threadIdx.x;
    if (idx >= N * NFEAT) return;
    const int n = idx / NFEAT, col = idx % NFEAT;
    const int grp = col / NWAVE, w = col % NWAVE;
    const float* Bn = &B[(size_t)n * NL * NWAVE];
    float s = 0.f;
    if (grp == 0) { const float v = Bn[w]; s = v * v; }
    else if (grp == 1) {
        for (int l = 1; l < 4; ++l) { const float v = Bn[l * NWAVE + w]; s += v * v; }
    } else {
        for (int l = 4; l < 13; ++l) { const float v = Bn[l * NWAVE + w]; s += v * v; }
    }
    dens[idx] = s;
}

__global__ void eann_dB(const float* __restrict__ B, const float* __restrict__ ddens,
                        float* __restrict__ dB, int N)
{
    const int idx = blockIdx.x * blockDim.x + threadIdx.x;
    if (idx >= N * NL * NWAVE) return;
    const int n = idx / (NL * NWAVE);
    const int r = idx % (NL * NWAVE);
    const int l = r / NWAVE, w = r % NWAVE;
    const int grp = (l == 0) ? 0 : ((l < 4) ? 1 : 2);
    dB[idx] = 2.f * B[idx] * ddens[(size_t)n * NFEAT + grp * NWAVE + w];
}

// dE/dh4 seed: wout broadcast on rows matching species t, else 0.
__global__ void eann_dh4_init(float* __restrict__ dh, const float* __restrict__ wout,
                              const int* __restrict__ species, int t, int N, int O)
{
    const int idx = blockIdx.x * blockDim.x + threadIdx.x;
    if (idx >= N * O) return;
    const int n = idx / O, o = idx % O;
    dh[idx] = (species[n] == t) ? wout[o] : 0.f;
}

// dz = dh * a*g*silu'(g*z)   (in place)
__global__ void eann_actgrad(float* __restrict__ dh, const float* __restrict__ z,
                             const float* __restrict__ a, const float* __restrict__ g,
                             int total, int O)
{
    const int idx = blockIdx.x * blockDim.x + threadIdx.x;
    if (idx >= total) return;
    const int o = idx % O;
    const float gz = g[o] * z[idx];
    const float s = sigf(gz);
    dh[idx] *= a[o] * g[o] * (s * (1.f + gz * (1.f - s)));
}

// e2 accumulation from stored pre-activations of net-2 layer 4.
__global__ void eann_e2(const float* __restrict__ z4, const float* __restrict__ a4,
                        const float* __restrict__ g4, const float* __restrict__ wout,
                        const float* __restrict__ bout, const int* __restrict__ species,
                        int t, float* __restrict__ acc, int N)
{
    const int n = blockIdx.x * blockDim.x + threadIdx.x;
    if (n >= N || species[n] != t) return;
    float s = bout[0];
#pragma unroll
    for (int o = 0; o < 32; ++o) {
        const float gz = g4[o] * z4[(size_t)n * 32 + o];
        s += (a4[o] * gz * sigf(gz)) * wout[o];
    }
    atomicAdd(acc, s);
}

// Pair backward: recompute geometry, chain rule through ang/gauss/fc, scatter.
__global__ void eann_pair_backward(const float* __restrict__ cart,
                                   const float* __restrict__ shifts,
                                   const float* __restrict__ rs,
                                   const float* __restrict__ inta,
                                   const float* __restrict__ cpar,
                                   const int* __restrict__ species,
                                   const int* __restrict__ neigh,
                                   const float* __restrict__ dB,
                                   int M, float* __restrict__ jab)
{
    const int m = blockIdx.x * blockDim.x + threadIdx.x;
    if (m >= M) return;
    const int i = neigh[m];
    const int j = neigh[M + m];
    float dv[3], d2 = 0.f;
#pragma unroll
    for (int c = 0; c < 3; ++c) {
        dv[c] = cart[i * 3 + c] - cart[j * 3 + c] + shifts[m * 3 + c];
        d2 += dv[c] * dv[c];
    }
    const float dist = sqrtf(fmaxf(d2, 1e-24f));
    const float th = dist * PI_OVER_CUT;
    const float cc = 0.5f * __cosf(th) + 0.5f;
    const float fc = cc * cc;
    const float dfc_ddist = 2.f * cc * (-0.5f * __sinf(th)) * PI_OVER_CUT;

    float ang[NL];
    ang[0] = fc;
#pragma unroll
    for (int c = 0; c < 3; ++c) ang[1 + c] = fc * dv[c];
#pragma unroll
    for (int a = 0; a < 3; ++a)
#pragma unroll
        for (int b = 0; b < 3; ++b) ang[4 + 3 * a + b] = fc * dv[a] * dv[b];

    const int sj = species[j];
    const float* dBi = &dB[(size_t)i * NL * NWAVE];
    float dAng[NL];
#pragma unroll
    for (int l = 0; l < NL; ++l) dAng[l] = 0.f;
    float gdist = 0.f;
#pragma unroll
    for (int w = 0; w < NWAVE; ++w) {
        const float dr = dist - rs[sj * NWAVE + w];
        const float ia = inta[sj * NWAVE + w];
        const float gw = __expf(ia * dr * dr) * cpar[sj * NWAVE + w];
        const float dgw = gw * 2.f * ia * dr;
        float dEdg = 0.f;
#pragma unroll
        for (int l = 0; l < NL; ++l) {
            const float db = dBi[l * NWAVE + w];
            dAng[l] += db * gw;
            dEdg    += db * ang[l];
        }
        gdist += dEdg * dgw;
    }
    float dfc = dAng[0];
    float gv[3] = {0.f, 0.f, 0.f};
#pragma unroll
    for (int c = 0; c < 3; ++c) { dfc += dAng[1 + c] * dv[c]; gv[c] += dAng[1 + c] * fc; }
#pragma unroll
    for (int a = 0; a < 3; ++a)
#pragma unroll
        for (int b = 0; b < 3; ++b) {
            const float d = dAng[4 + 3 * a + b];
            dfc += d * dv[a] * dv[b];
            gv[a] += d * fc * dv[b];
            gv[b] += d * fc * dv[a];
        }
    gdist += dfc * dfc_ddist;
    const float inv = __builtin_amdgcn_rcpf(dist);
#pragma unroll
    for (int c = 0; c < 3; ++c) {
        const float gt = gv[c] + gdist * dv[c] * inv;
        atomicAdd(&jab[i * 3 + c],  gt);
        atomicAdd(&jab[j * 3 + c], -gt);
    }
}

// acc[0..8]  += sum_n cart[n,j]*jab1[n,k] ; acc[9..17] += sum_n jab2[n,j]*jab2[n,k]
__global__ void eann_polar_reduce(const float* __restrict__ cart,
                                  const float* __restrict__ jab1,
                                  const float* __restrict__ jab2,
                                  float* __restrict__ acc, int N)
{
    float pa[9], pb[9];
#pragma unroll
    for (int q = 0; q < 9; ++q) { pa[q] = 0.f; pb[q] = 0.f; }
    for (int n = blockIdx.x * blockDim.x + threadIdx.x; n < N; n += gridDim.x * blockDim.x) {
#pragma unroll
        for (int jj = 0; jj < 3; ++jj)
#pragma unroll
            for (int kk = 0; kk < 3; ++kk) {
                pa[jj * 3 + kk] += cart[n * 3 + jj] * jab1[n * 3 + kk];
                pb[jj * 3 + kk] += jab2[n * 3 + jj] * jab2[n * 3 + kk];
            }
    }
#pragma unroll
    for (int q = 0; q < 9; ++q) {
        atomicAdd(&acc[q],     pa[q]);
        atomicAdd(&acc[9 + q], pb[q]);
    }
}

__global__ void eann_finalize(const float* __restrict__ acc, float* __restrict__ out)
{
    const int idx = threadIdx.x;
    if (idx < 9) {
        const int jj = idx / 3, kk = idx % 3;
        float v = acc[jj * 3 + kk] + acc[kk * 3 + jj] + acc[9 + idx];
        if (jj == kk) v += acc[18];
        out[idx] = v;
    }
}

// ---------------------------------------------------------------------------
// Host-side GEMM dispatch (shapes fixed by the reference's DIMS).
// ---------------------------------------------------------------------------
static void launch_gemm(const float* X, const float* aA, const float* gA,
                        const float* W, const float* bias, float* Y,
                        int K, int O, bool transb, int accum, int N, hipStream_t s)
{
#define EG(KK, OO, CC, TB, HA)                                                     \
    {                                                                              \
        const dim3 g(N / 16, ((OO) + (CC)*16 - 1) / ((CC)*16));                    \
        eann_gemm_t<KK, OO, CC, TB, HA><<<g, 32, 0, s>>>(X, aA, gA, W, bias, Y, accum); \
        return;                                                                    \
    }
    if (!transb) {
        if (K == 36  && O == 256) EG(36, 256, 4, false, false)
        if (K == 256 && O == 128) EG(256, 128, 4, false, true)
        if (K == 128 && O == 64)  EG(128, 64, 4, false, true)
        if (K == 64  && O == 32)  EG(64, 32, 2, false, true)
    } else {
        if (K == 32  && O == 64)  EG(32, 64, 4, true, false)
        if (K == 64  && O == 128) EG(64, 128, 4, true, false)
        if (K == 128 && O == 256) EG(128, 256, 4, true, false)
        if (K == 256 && O == 36)  EG(256, 36, 3, true, false)
    }
#undef EG
}

// ---------------------------------------------------------------------------
extern "C" void kernel_launch(void* const* d_in, const int* in_sizes, int n_in,
                              void* d_out, int out_size, void* d_ws, size_t ws_size,
                              hipStream_t stream)
{
    const float* cart   = (const float*)d_in[0];
    const float* shifts = (const float*)d_in[1];
    const float* rs     = (const float*)d_in[2];
    const float* inta   = (const float*)d_in[3];
    const float* cpar   = (const float*)d_in[4];
    const float* wL[4]  = {(const float*)d_in[5],  (const float*)d_in[9],
                           (const float*)d_in[13], (const float*)d_in[17]};
    const float* bL[4]  = {(const float*)d_in[6],  (const float*)d_in[10],
                           (const float*)d_in[14], (const float*)d_in[18]};
    const float* aL[4]  = {(const float*)d_in[7],  (const float*)d_in[11],
                           (const float*)d_in[15], (const float*)d_in[19]};
    const float* gL[4]  = {(const float*)d_in[8],  (const float*)d_in[12],
                           (const float*)d_in[16], (const float*)d_in[20]};
    const float* wout   = (const float*)d_in[21];
    const float* bout   = (const float*)d_in[22];
    const int*   species = (const int*)d_in[23];
    const int*   neigh   = (const int*)d_in[24];

    const int N  = in_sizes[0] / 3;          // atoms (8192)
    const int M  = in_sizes[1] / 3;          // pairs (393216)
    int KN = M / N;                          // neighbors per center (48)
    if (KN > 48) KN = 48;

    static const int fdims[5] = {36, 256, 128, 64, 32};
    const int zoff[4] = {0, N * 256, N * 384, N * 448};

    // ---- scratch layout (floats) ----
    float* ws    = (float*)d_ws;
    float* B     = ws;                           // N*156
    float* dB    = B     + (size_t)N * 156;      // N*156
    float* dens  = dB    + (size_t)N * 156;      // N*36
    float* ddens = dens  + (size_t)N * 36;       // N*36
    float* z     = ddens + (size_t)N * 36;       // N*480
    float* dhA   = z     + (size_t)N * 480;      // N*256
    float* dhB   = dhA   + (size_t)N * 256;      // N*256
    float* jab1  = dhB   + (size_t)N * 256;      // N*3
    float* jab2  = jab1  + (size_t)N * 3;        // N*3
    float* acc   = jab2  + (size_t)N * 3;        // 19 (A[9], J2[9], e2)

    hipMemsetAsync(acc,  0, 19 * sizeof(float), stream);
    hipMemsetAsync(jab1, 0, (size_t)N * 3 * sizeof(float), stream);
    hipMemsetAsync(jab2, 0, (size_t)N * 3 * sizeof(float), stream);

    // ---- density forward ----
    eann_pair_forward<<<N, 192, 0, stream>>>(cart, shifts, rs, inta, cpar,
                                             species, neigh, M, KN, B);
    eann_density<<<(N * NFEAT + 255) / 256, 256, 0, stream>>>(B, dens, N);

    // ---- nets 0,1: forward + backward -> jab1/jab2 ----
    for (int nnet = 0; nnet < 2; ++nnet) {
        float* jab = (nnet == 0) ? jab1 : jab2;
        hipMemsetAsync(ddens, 0, (size_t)N * NFEAT * sizeof(float), stream);
        for (int t = 0; t < 2; ++t) {
            const int nt = nnet * 2 + t;
            // forward: store pre-activations z1..z4 (activation fused on consume)
            const float* Xp = dens;
            const float *aAp = nullptr, *gAp = nullptr;
            for (int L = 0; L < 4; ++L) {
                const int fi = fdims[L], fo = fdims[L + 1];
                launch_gemm(Xp, aAp, gAp,
                            wL[L] + (size_t)nt * fi * fo,
                            bL[L] + (size_t)nt * fo,
                            z + zoff[L], fi, fo, false, 0, N, stream);
                Xp = z + zoff[L];
                aAp = aL[L] + (size_t)nt * fo;
                gAp = gL[L] + (size_t)nt * fo;
            }
            // backward: dh4 seed -> chain to ddens (accumulated over types)
            eann_dh4_init<<<(N * 32 + 255) / 256, 256, 0, stream>>>(
                dhA, wout + (size_t)nt * 32, species, t, N, 32);
            float* cur = dhA;
            for (int L = 3; L >= 0; --L) {
                const int fi = fdims[L], fo = fdims[L + 1];
                eann_actgrad<<<(N * fo + 255) / 256, 256, 0, stream>>>(
                    cur, z + zoff[L], aL[L] + (size_t)nt * fo, gL[L] + (size_t)nt * fo,
                    N * fo, fo);
                float* nxt = (L > 0) ? ((cur == dhA) ? dhB : dhA) : ddens;
                launch_gemm(cur, nullptr, nullptr,
                            wL[L] + (size_t)nt * fi * fo, nullptr,
                            nxt, fo, fi, true, (L == 0) ? 1 : 0, N, stream);
                cur = nxt;
            }
        }
        eann_dB<<<(N * NL * NWAVE + 255) / 256, 256, 0, stream>>>(B, ddens, dB, N);
        eann_pair_backward<<<(M + 255) / 256, 256, 0, stream>>>(
            cart, shifts, rs, inta, cpar, species, neigh, dB, M, jab);
    }

    // ---- net 2: forward only, accumulate e2 ----
    for (int t = 0; t < 2; ++t) {
        const int nt = 2 * 2 + t;
        const float* Xp = dens;
        const float *aAp = nullptr, *gAp = nullptr;
        for (int L = 0; L < 4; ++L) {
            const int fi = fdims[L], fo = fdims[L + 1];
            launch_gemm(Xp, aAp, gAp,
                        wL[L] + (size_t)nt * fi * fo,
                        bL[L] + (size_t)nt * fo,
                        z + zoff[L], fi, fo, false, 0, N, stream);
            Xp = z + zoff[L];
            aAp = aL[L] + (size_t)nt * fo;
            gAp = gL[L] + (size_t)nt * fo;
        }
        eann_e2<<<(N + 255) / 256, 256, 0, stream>>>(
            z + zoff[3], aL[3] + (size_t)nt * 32, gL[3] + (size_t)nt * 32,
            wout + (size_t)nt * 32, bout + nt, species, t, acc + 18, N);
    }

    // ---- polar assembly ----
    eann_polar_reduce<<<64, 256, 0, stream>>>(cart, jab1, jab2, acc, N);
    eann_finalize<<<1, 32, 0, stream>>>(acc, (float*)d_out);
}
